// BiLSTM_Parser_21440476741954
// MI455X (gfx1250) — matrix operand order repaired and checked by reading, not compiled, not run on hardware
//
#include <hip/hip_runtime.h>
#include <hip/hip_bf16.h>
#include <math.h>

// ---------------------------------------------------------------------------
// BiLSTM parser for MI455X (gfx1250, wave32).
//  - Dense GEMMs: v_wmma_f32_16x16x32_bf16, 32x64 wave tiles, software-
//    pipelined fragment loads (prefetch next-k while current WMMAs run).
//  - Recurrent scan: 32 WGs/direction; each thread keeps its 128 Whh weights
//    in VGPRs for the whole scan; h broadcast through padded LDS; step
//    barrier via agent-scope atomics (release/acquire) + s_sleep.
// Dims: L=1024, H=512/dir, 4H=2048, IN0=400(pad 512), IN1=1024, M/2=512.
// ---------------------------------------------------------------------------

typedef __attribute__((ext_vector_type(16))) __bf16 v16bf;
typedef __attribute__((ext_vector_type(8)))  float  v8f;

#define NSPLIT 32   // workgroups per direction in the scan

// ---------------- weight convert (f32 -> bf16, K-padded) -------------------
__global__ void cvt_pad_bf16(const float* __restrict__ in, __bf16* __restrict__ out,
                             int R, int Kin, int Kpad) {
  size_t total = (size_t)R * Kpad;
  for (size_t i = (size_t)blockIdx.x * blockDim.x + threadIdx.x; i < total;
       i += (size_t)gridDim.x * blockDim.x) {
    int r = (int)(i / Kpad), c = (int)(i % Kpad);
    out[i] = (c < Kin) ? (__bf16)in[(size_t)r * Kin + c] : (__bf16)0.f;
  }
}

// ---------------- embedding gather -> bf16 [1024][512] ---------------------
__global__ void embed_kernel(const int* __restrict__ wt, const int* __restrict__ pt,
                             const float* __restrict__ wemb, const float* __restrict__ pemb,
                             __bf16* __restrict__ xb) {
  int t = blockIdx.x;
  int w = wt[t], p = pt[t];
  for (int c = threadIdx.x; c < 512; c += blockDim.x) {
    float v = (c < 300) ? wemb[(size_t)w * 300 + c]
                        : ((c < 400) ? pemb[(size_t)p * 100 + (c - 300)] : 0.f);
    xb[(size_t)t * 512 + c] = (__bf16)v;
  }
}

// ---------------- WMMA bf16 GEMM: C[M,N] = A[M,K] * W[N,K]^T + bias --------
// A row-major (lda), W row-major over K (ldb).  One wave = 32(M) x 64(N).
// Fragment layouts per CDNA5 ISA 7.12.2 (16-bit A 16x32, B 32x16).
__device__ inline v16bf load_a_frag(const __bf16* A, int lda, int m0, int k0, int lane) {
  int r = lane & 15, hi = lane >> 4;
  const __bf16* p = A + (size_t)(m0 + r) * lda + k0 + hi * 8;
  v16bf a;
  uint4* d = reinterpret_cast<uint4*>(&a);
  d[0] = *reinterpret_cast<const uint4*>(p);        // K = hi*8 .. hi*8+7
  d[1] = *reinterpret_cast<const uint4*>(p + 16);   // K = 16+hi*8 ..
  return a;
}
__device__ inline v16bf load_b_frag(const __bf16* W, int ldb, int n0, int k0, int lane) {
  int c = lane & 15, kh = lane >> 4;
  const __bf16* p = W + (size_t)(n0 + c) * ldb + k0 + kh * 8;    // K = kh*8.., 16+kh*8..
  v16bf b;
  uint4* d = reinterpret_cast<uint4*>(&b);
  d[0] = *reinterpret_cast<const uint4*>(p);
  d[1] = *reinterpret_cast<const uint4*>(p + 16);
  return b;
}

__global__ void __launch_bounds__(256)
gemm_bf16_wmma(const __bf16* __restrict__ A, int lda,
               const __bf16* __restrict__ W, int ldb,
               const float* __restrict__ b1, const float* __restrict__ b2,
               float* __restrict__ C, int ldc,
               int Mtiles /*M/32*/, int Ntiles /*N/64*/, int K) {
  int wave = blockIdx.x * (blockDim.x >> 5) + (threadIdx.x >> 5);
  if (wave >= Mtiles * Ntiles) return;
  int lane = threadIdx.x & 31;
  int m0 = (wave / Ntiles) * 32;
  int n0 = (wave % Ntiles) * 64;

  const v8f vzero = {0.f, 0.f, 0.f, 0.f, 0.f, 0.f, 0.f, 0.f};
  v8f acc[8];
#pragma unroll
  for (int s = 0; s < 8; ++s) acc[s] = vzero;

  // Prologue fragments (k0 = 0).
  v16bf a0 = load_a_frag(A, lda, m0,      0, lane);
  v16bf a1 = load_a_frag(A, lda, m0 + 16, 0, lane);
  v16bf b0 = load_b_frag(W, ldb, n0,      0, lane);
  v16bf b1v = load_b_frag(W, ldb, n0 + 16, 0, lane);
  v16bf b2v = load_b_frag(W, ldb, n0 + 32, 0, lane);
  v16bf b3 = load_b_frag(W, ldb, n0 + 48, 0, lane);

  for (int k0 = 0; k0 < K; k0 += 32) {
    // Branchless prefetch of the next k-slice (re-reads last slice on epilogue).
    int kn = (k0 + 32 < K) ? (k0 + 32) : k0;
    v16bf na0 = load_a_frag(A, lda, m0,      kn, lane);
    v16bf na1 = load_a_frag(A, lda, m0 + 16, kn, lane);
    v16bf nb0 = load_b_frag(W, ldb, n0,      kn, lane);
    v16bf nb1 = load_b_frag(W, ldb, n0 + 16, kn, lane);
    v16bf nb2 = load_b_frag(W, ldb, n0 + 32, kn, lane);
    v16bf nb3 = load_b_frag(W, ldb, n0 + 48, kn, lane);

    acc[0] = __builtin_amdgcn_wmma_f32_16x16x32_bf16(false, a0, false, b0,  (short)0, acc[0], false, false);
    acc[1] = __builtin_amdgcn_wmma_f32_16x16x32_bf16(false, a0, false, b1v, (short)0, acc[1], false, false);
    acc[2] = __builtin_amdgcn_wmma_f32_16x16x32_bf16(false, a0, false, b2v, (short)0, acc[2], false, false);
    acc[3] = __builtin_amdgcn_wmma_f32_16x16x32_bf16(false, a0, false, b3,  (short)0, acc[3], false, false);
    acc[4] = __builtin_amdgcn_wmma_f32_16x16x32_bf16(false, a1, false, b0,  (short)0, acc[4], false, false);
    acc[5] = __builtin_amdgcn_wmma_f32_16x16x32_bf16(false, a1, false, b1v, (short)0, acc[5], false, false);
    acc[6] = __builtin_amdgcn_wmma_f32_16x16x32_bf16(false, a1, false, b2v, (short)0, acc[6], false, false);
    acc[7] = __builtin_amdgcn_wmma_f32_16x16x32_bf16(false, a1, false, b3,  (short)0, acc[7], false, false);

    a0 = na0; a1 = na1; b0 = nb0; b1v = nb1; b2v = nb2; b3 = nb3;
  }

  // C/D layout: lane<16 -> N=lane, M=v ; lane>=16 -> N=lane-16, M=8+v
  int r = lane & 15, top = lane >> 4;
#pragma unroll
  for (int half = 0; half < 2; ++half) {
#pragma unroll
    for (int s = 0; s < 4; ++s) {
      int col = n0 + s * 16 + r;
      float bias = (b1 ? b1[col] : 0.f) + (b2 ? b2[col] : 0.f);
#pragma unroll
      for (int v = 0; v < 8; ++v)
        C[(size_t)(m0 + half * 16 + v + 8 * top) * ldc + col] = acc[half * 4 + s][v] + bias;
    }
  }
}

// ---------------- zero the scan sync state ---------------------------------
__global__ void init_sync(float* hbuf, unsigned* cnt) {
  int i = blockIdx.x * blockDim.x + threadIdx.x;
  if (i < 2 * 2 * 512) hbuf[i] = 0.f;
  if (i < 2) cnt[i] = 0u;
}

// ---------------- recurrent scan (one LSTM layer, both directions) ---------
// grid = 64 WGs: blockIdx/32 = direction, blockIdx%32 = hidden-slice k.
// Each WG owns 16 hidden units -> 64 gate rows of Whh.  Each thread keeps its
// 128 weights (gate row `row`, k = q*128 .. q*128+127) in VGPRs for the whole scan; only
// the h vector moves through LDS (padded stride 132 -> conflict-free b128
// reads, 8-lane same-address broadcast).  Cross-WG step barrier: per-direction
// counter, release/acquire, s_sleep spin.
__global__ void __launch_bounds__(256)
lstm_scan(const float* __restrict__ Whh,    // [2][2048][512] f32 (input tensor)
          const float* __restrict__ pre,    // [2][1024][2048] f32 (x-proj + biases)
          __bf16* __restrict__ houtb,       // [1024][1024] bf16 (fwd 0..511, bwd 512..1023)
          float* hbuf,                      // [2][2][512] f32 double-buffered h
          unsigned* cnt) {                  // [2] step counters
  __shared__ float hsh[4 * 132];            // h, padded: k -> (k>>7)*132 + (k&127)
  __shared__ float gred[64];
  __shared__ float c_s[16];

  const int tid = threadIdx.x;
  const int dir = blockIdx.x >> 5;
  const int k   = blockIdx.x & (NSPLIT - 1);
  const int j0  = k * 16;

  const int row = tid >> 2;                 // gate row 0..63 within this WG
  const int q   = tid & 3;                  // k-quad lane
  const int grow = (row >> 4) * 512 + j0 + (row & 15);

  // One-time fill: this thread's 128 Whh weights (k = q*128 .. q*128+127,
  // contiguous -> coalesced float4 loads) stay in VGPRs for all 1024 steps.
  float4 wreg[32];
  {
    const float4* wsrc = reinterpret_cast<const float4*>(
        Whh + ((size_t)dir * 2048 + grow) * 512 + q * 128);
#pragma unroll
    for (int i = 0; i < 32; ++i) wreg[i] = wsrc[i];
  }
  if (tid < 16) c_s[tid] = 0.f;
  __syncthreads();

  unsigned* mycnt = cnt + dir;

  for (int t = 0; t < 1024; ++t) {
    // Wait until every WG of this direction finished step t-1.
    if (tid == 0) {
      unsigned tgt = (unsigned)(NSPLIT * t);
      while (__hip_atomic_load(mycnt, __ATOMIC_ACQUIRE, __HIP_MEMORY_SCOPE_AGENT) < tgt)
        __builtin_amdgcn_s_sleep(1);
    }
    __syncthreads();

    const float* hprev = hbuf + ((size_t)dir * 2 + ((t ^ 1) & 1)) * 512;
    for (int c = tid; c < 512; c += 256)
      hsh[(c >> 7) * 132 + (c & 127)] = hprev[c];
    __syncthreads();

    // dot(Whh[grow], h): this thread covers k = q*128 .. q*128+127.
    float sum = 0.f;
    const float4* hq = reinterpret_cast<const float4*>(&hsh[q * 132]);
#pragma unroll
    for (int i = 0; i < 32; ++i) {
      float4 h4 = hq[i];
      float4 w4 = wreg[i];
      sum += w4.x * h4.x + w4.y * h4.y + w4.z * h4.z + w4.w * h4.w;
    }
    sum += __shfl_xor(sum, 1);
    sum += __shfl_xor(sum, 2);
    if (q == 0) gred[row] = sum;
    __syncthreads();

    const int tact = dir ? (1023 - t) : t;
    if (tid < 16) {
      const float* pt = pre + ((size_t)dir * 1024 + tact) * 2048;
      int j = j0 + tid;
      float gi = pt[j]        + gred[tid];
      float gf = pt[512 + j]  + gred[16 + tid];
      float gg = pt[1024 + j] + gred[32 + tid];
      float go = pt[1536 + j] + gred[48 + tid];
      float si = 1.f / (1.f + __expf(-gi));
      float sf = 1.f / (1.f + __expf(-gf));
      float so = 1.f / (1.f + __expf(-go));
      float c  = sf * c_s[tid] + si * tanhf(gg);
      c_s[tid] = c;
      float h = so * tanhf(c);
      hbuf[((size_t)dir * 2 + (t & 1)) * 512 + j] = h;
      houtb[(size_t)tact * 1024 + dir * 512 + j] = (__bf16)h;
      __threadfence();                     // release our h chunk to the agent
    }
    __syncthreads();
    if (tid == 0)
      __hip_atomic_fetch_add(mycnt, 1u, __ATOMIC_RELEASE, __HIP_MEMORY_SCOPE_AGENT);
  }
}

// ---------------- head/modif tanh-dot reduction ----------------------------
__global__ void __launch_bounds__(256)
head_reduce(const float* __restrict__ headf, const float* __restrict__ modiff,
            const float* __restrict__ Wout, float* __restrict__ hs, float* __restrict__ ms) {
  __shared__ float s1[256], s2[256];
  int t = blockIdx.x, tid = threadIdx.x;
  float a = 0.f, b = 0.f;
  for (int n = tid; n < 512; n += 256) {
    a += tanhf(headf[(size_t)t * 512 + n]) * Wout[n];
    b += tanhf(modiff[(size_t)t * 512 + n]) * Wout[512 + n];
  }
  s1[tid] = a; s2[tid] = b;
  __syncthreads();
  for (int s = 128; s > 0; s >>= 1) {
    if (tid < s) { s1[tid] += s1[tid + s]; s2[tid] += s2[tid + s]; }
    __syncthreads();
  }
  if (tid == 0) { hs[t] = s1[0]; ms[t] = s2[0]; }
}

// ---------------- final outer sum ------------------------------------------
__global__ void outer_kernel(const float* __restrict__ hs, const float* __restrict__ ms,
                             const float* __restrict__ bout, float* __restrict__ out) {
  int i = blockIdx.x * blockDim.x + threadIdx.x;
  if (i < 1024 * 1024) {
    int m = i >> 10, h = i & 1023;
    out[i] = ms[m] + hs[h] + bout[0];
  }
}

// ---------------------------------------------------------------------------
extern "C" void kernel_launch(void* const* d_in, const int* in_sizes, int n_in,
                              void* d_out, int out_size, void* d_ws, size_t ws_size,
                              hipStream_t stream) {
  (void)in_sizes; (void)n_in; (void)out_size; (void)ws_size;
  const int*   wt    = (const int*)  d_in[0];
  const int*   pt    = (const int*)  d_in[1];
  const float* wemb  = (const float*)d_in[2];
  const float* pemb  = (const float*)d_in[3];
  const float* Wih0  = (const float*)d_in[4];
  const float* Whh0  = (const float*)d_in[5];
  const float* bih0  = (const float*)d_in[6];
  const float* bhh0  = (const float*)d_in[7];
  const float* Wih1  = (const float*)d_in[8];
  const float* Whh1  = (const float*)d_in[9];
  const float* bih1  = (const float*)d_in[10];
  const float* bhh1  = (const float*)d_in[11];
  const float* Whead = (const float*)d_in[12];
  const float* bhead = (const float*)d_in[13];
  const float* Wmod  = (const float*)d_in[14];
  const float* bmod  = (const float*)d_in[15];
  const float* Wout  = (const float*)d_in[16];
  const float* bout  = (const float*)d_in[17];
  float* out = (float*)d_out;

  // Workspace carve (~41 MB total), 256B aligned.
  char* base = (char*)d_ws;
  size_t o = 0;
  auto carve = [&](size_t bytes) -> char* {
    char* p = base + o;
    o = (o + bytes + 255) & ~(size_t)255;
    return p;
  };
  __bf16* xb    = (__bf16*)carve((size_t)1024 * 512 * 2);        // padded input
  __bf16* Wb0   = (__bf16*)carve((size_t)2 * 2048 * 512 * 2);    // Wih_l0 bf16 (K pad 512)
  __bf16* Wb1   = (__bf16*)carve((size_t)2 * 2048 * 1024 * 2);   // Wih_l1 bf16
  __bf16* Whd   = (__bf16*)carve((size_t)512 * 1024 * 2);        // W_head bf16
  __bf16* Wmd   = (__bf16*)carve((size_t)512 * 1024 * 2);        // W_modif bf16
  float*  preb  = (float*) carve((size_t)2 * 1024 * 2048 * 4);   // gate pre-acts (reused L0/L1)
  __bf16* h0b   = (__bf16*)carve((size_t)1024 * 1024 * 2);       // layer-0 output
  __bf16* h1b   = (__bf16*)carve((size_t)1024 * 1024 * 2);       // layer-1 output
  float*  headf = (float*) carve((size_t)1024 * 512 * 4);
  float*  modif = (float*) carve((size_t)1024 * 512 * 4);
  float*  hsv   = (float*) carve((size_t)1024 * 4);
  float*  msv   = (float*) carve((size_t)1024 * 4);
  float*  hbuf  = (float*) carve((size_t)2 * 2 * 512 * 4);
  unsigned* cnt = (unsigned*)carve(256);

  // 1) weight conversions + embedding gather
  cvt_pad_bf16<<<2048, 256, 0, stream>>>(Wih0, Wb0, 4096, 400, 512);
  cvt_pad_bf16<<<4096, 256, 0, stream>>>(Wih1, Wb1, 4096, 1024, 1024);
  cvt_pad_bf16<<<1024, 256, 0, stream>>>(Whead, Whd, 512, 1024, 1024);
  cvt_pad_bf16<<<1024, 256, 0, stream>>>(Wmod,  Wmd, 512, 1024, 1024);
  embed_kernel<<<1024, 128, 0, stream>>>(wt, pt, wemb, pemb, xb);

  // 2) layer-0 input projection: pre[d] = xb @ Wih0[d]^T + (bih0+bhh0)[d]
  for (int d = 0; d < 2; ++d)
    gemm_bf16_wmma<<<128, 256, 0, stream>>>(
        xb, 512, Wb0 + (size_t)d * 2048 * 512, 512,
        bih0 + d * 2048, bhh0 + d * 2048,
        preb + (size_t)d * 1024 * 2048, 2048, 32, 32, 512);

  // 3) layer-0 recurrent scan
  init_sync<<<8, 256, 0, stream>>>(hbuf, cnt);
  lstm_scan<<<2 * NSPLIT, 256, 0, stream>>>(Whh0, preb, h0b, hbuf, cnt);

  // 4) layer-1 input projection from h0
  for (int d = 0; d < 2; ++d)
    gemm_bf16_wmma<<<128, 256, 0, stream>>>(
        h0b, 1024, Wb1 + (size_t)d * 2048 * 1024, 1024,
        bih1 + d * 2048, bhh1 + d * 2048,
        preb + (size_t)d * 1024 * 2048, 2048, 32, 32, 1024);

  // 5) layer-1 recurrent scan
  init_sync<<<8, 256, 0, stream>>>(hbuf, cnt);
  lstm_scan<<<2 * NSPLIT, 256, 0, stream>>>(Whh1, preb, h1b, hbuf, cnt);

  // 6) head / modifier projections
  gemm_bf16_wmma<<<32, 256, 0, stream>>>(h1b, 1024, Whd, 1024, bhead, nullptr,
                                         headf, 512, 32, 8, 1024);
  gemm_bf16_wmma<<<32, 256, 0, stream>>>(h1b, 1024, Wmd, 1024, bmod, nullptr,
                                         modif, 512, 32, 8, 1024);

  // 7) tanh-dot reductions + outer sum -> score matrix
  head_reduce<<<1024, 256, 0, stream>>>(headf, modif, Wout, hsv, msv);
  outer_kernel<<<4096, 256, 0, stream>>>(hsv, msv, bout, out);
}
